// EarthAttention2D_76390288327291
// MI455X (gfx1250) — matrix operand order repaired and checked by reading, not compiled, not run on hardware
//
#include <hip/hip_runtime.h>
#include <hip/hip_bf16.h>

// ---- problem constants (fixed by the reference) ----
#define B_SZ  16
#define NW    60
#define NH    12
#define NTK   72      // window tokens (6*12)
#define CD    384
#define HD    32
#define O3    1152    // 3*C
#define TTOT  (B_SZ*NW*NTK)   // 69120 tokens

typedef __attribute__((ext_vector_type(16))) __bf16 bf16x16;
typedef __attribute__((ext_vector_type(8)))  float  f32x8;

#if defined(__gfx1250__) && \
    __has_builtin(__builtin_amdgcn_global_load_async_to_lds_b128) && \
    __has_builtin(__builtin_amdgcn_s_wait_asynccnt)
#define HAVE_ASYNC_LDS 1
#else
#define HAVE_ASYNC_LDS 0
#endif

#if HAVE_ASYNC_LDS
// Builtin signature (from hipcc diagnostic): param0 = AS1 pointer to
// GCC-vector int4, param1 = AS3 pointer, then imm offset + imm cpol.
typedef int v4i_vs __attribute__((vector_size(4 * sizeof(int))));
typedef __attribute__((address_space(1))) v4i_vs* gptr_v4i;
typedef __attribute__((address_space(3))) v4i_vs* lptr_v4i;

// Per-lane 16B global->LDS async copy (ASYNCcnt-tracked, no VGPR round trip).
// AS pointers built via integer round trip: AS3 offset == low 32 bits of the
// flat LDS address (aperture layout, ISA 10.2).
__device__ __forceinline__ void async_cp16(const void* g, void* l) {
  __builtin_amdgcn_global_load_async_to_lds_b128(
      (gptr_v4i)(uintptr_t)g,
      (lptr_v4i)(unsigned int)(uintptr_t)l,
      0, 0);
}
#endif

// Load a 16-element bf16 fragment as two 16B LDS reads (ds_load_b128 x2).
__device__ __forceinline__ bf16x16 ld_frag2(const __bf16* p0, const __bf16* p1) {
  bf16x16 r;
  ((float4*)&r)[0] = *(const float4*)p0;
  ((float4*)&r)[1] = *(const float4*)p1;
  return r;
}

__device__ __forceinline__ f32x8 wmma_bf16(bf16x16 a, bf16x16 b, f32x8 c) {
  return __builtin_amdgcn_wmma_f32_16x16x32_bf16(false, a, false, b, (short)0, c,
                                                 false, false);
}

// ------------------------------------------------------------------
__global__ void cvt_f32_to_bf16(const float* __restrict__ s,
                                __bf16* __restrict__ d, int n) {
  int i = blockIdx.x * 256 + threadIdx.x;
  if (i < n) d[i] = (__bf16)s[i];
}

// ------------------------------------------------------------------
// Tiled bf16 GEMM: C[M x Nout] = A[M x K=384] * W[Nout x K]^T + bias
// 128x128 block tile, K-step 32, double-buffered LDS, 8 waves (each 32x64).
// EPI==0: QKV epilogue (scatter bf16 q/k/v in (b,h,w,n,d), scale on q).
// EPI==1: proj epilogue (fp32 token-major out).
// ------------------------------------------------------------------
template<int EPI>
__global__ __launch_bounds__(256) void gemm_bf16_128(
    const __bf16* __restrict__ A, const __bf16* __restrict__ W,
    const float* __restrict__ bias,
    __bf16* __restrict__ qb, __bf16* __restrict__ kb, __bf16* __restrict__ vb,
    float* __restrict__ outp)
{
  __shared__ __align__(16) __bf16 As[2][128 * 32];
  __shared__ __align__(16) __bf16 Bs[2][128 * 32];   // [n][k], k contiguous

  const int tid  = threadIdx.x;
  const int lane = tid & 31, wave = tid >> 5;
  const int wr = wave >> 1, wc = wave & 1;           // 4x2 wave grid
  const int kh = lane >> 4, ln = lane & 15;
  const long m0 = (long)blockIdx.x * 128;
  const int  n0 = blockIdx.y * 128;
  const int lrow = tid >> 1, lh16 = (tid & 1) * 16;

  const f32x8 zero = {0.f,0.f,0.f,0.f,0.f,0.f,0.f,0.f};
  f32x8 acc[2][4];
  for (int i = 0; i < 2; i++)
    for (int j = 0; j < 4; j++) acc[i][j] = zero;

  const long arow = (m0 + lrow) * CD;
  const long wrow = (long)(n0 + lrow) * CD;
  const int lidx = lrow * 32 + lh16;

#if HAVE_ASYNC_LDS
  // prologue: async-stage tile 0
  async_cp16(&A[arow + lh16], &As[0][lidx]);
  async_cp16(&W[wrow + lh16], &Bs[0][lidx]);
  int cur = 0;
  for (int k0 = 0; k0 < CD; k0 += 32) {
    __builtin_amdgcn_s_wait_asynccnt(0);
    __syncthreads();
    if (k0 + 32 < CD) {          // overlap next-tile DMA with this tile's math
      async_cp16(&A[arow + k0 + 32 + lh16], &As[cur ^ 1][lidx]);
      async_cp16(&W[wrow + k0 + 32 + lh16], &Bs[cur ^ 1][lidx]);
    }
#else
  uint4 ra = *(const uint4*)&A[arow + lh16];
  uint4 rb = *(const uint4*)&W[wrow + lh16];
  int cur = 0;
  for (int k0 = 0; k0 < CD; k0 += 32) {
    *(uint4*)&As[cur][lidx] = ra;
    *(uint4*)&Bs[cur][lidx] = rb;
    __syncthreads();
    if (k0 + 32 < CD) {
      ra = *(const uint4*)&A[arow + k0 + 32 + lh16];
      rb = *(const uint4*)&W[wrow + k0 + 32 + lh16];
      __builtin_prefetch(&A[arow + k0 + 64 + lh16], 0, 0);
      __builtin_prefetch(&W[wrow + k0 + 64 + lh16], 0, 0);
    }
#endif
    bf16x16 af[2], bfrag[4];
    for (int mt = 0; mt < 2; mt++) {
      const __bf16* p = &As[cur][(wr * 32 + mt * 16 + ln) * 32];
      af[mt] = ld_frag2(p + 8 * kh, p + 16 + 8 * kh);
    }
    for (int nt = 0; nt < 4; nt++) {
      const __bf16* p = &Bs[cur][(wc * 64 + nt * 16 + ln) * 32];
      bfrag[nt] = ld_frag2(p + 8 * kh, p + 16 + 8 * kh);
    }
    for (int mt = 0; mt < 2; mt++)
      for (int nt = 0; nt < 4; nt++)
        acc[mt][nt] = wmma_bf16(af[mt], bfrag[nt], acc[mt][nt]);
    cur ^= 1;
  }

  // ---- epilogue: hoist all per-column decomposition out of inner loops ----
  float bv[4];
  if (EPI == 0) {
    long   colpart[4];
    float  scl[4];
    __bf16* dp[4];
    for (int nt = 0; nt < 4; nt++) {
      const int col = n0 + wc * 64 + nt * 16 + ln;
      bv[nt] = bias[col];
      const int s  = col / CD;           // 0:q 1:k 2:v
      const int rm = col % CD;
      const int hh = rm >> 5, dd = rm & 31;
      colpart[nt] = (long)hh * (NW * NTK * HD) + dd;   // 138240*hh + dd
      scl[nt] = (s == 0) ? 0.17677669529663687f : 1.f; // hd^-0.5 folded into q
      dp[nt]  = (s == 0) ? qb : ((s == 1) ? kb : vb);
    }
    for (int mt = 0; mt < 2; mt++)
      for (int r = 0; r < 8; r++) {
        const long row = m0 + wr * 32 + mt * 16 + kh * 8 + r;
        const int bi = (int)(row / (NW * NTK));
        const int rw = (int)(row % (NW * NTK));
        const int ww = rw / NTK, nn = rw % NTK;
        const long rowpart = (long)bi * (NH * NW * NTK * HD) +
                             (long)ww * (NTK * HD) + (long)nn * HD;
        for (int nt = 0; nt < 4; nt++)
          dp[nt][rowpart + colpart[nt]] =
              (__bf16)((acc[mt][nt][r] + bv[nt]) * scl[nt]);
      }
  } else {
    int ncol[4];
    for (int nt = 0; nt < 4; nt++) {
      ncol[nt] = n0 + wc * 64 + nt * 16 + ln;
      bv[nt] = bias[ncol[nt]];
    }
    for (int mt = 0; mt < 2; mt++)
      for (int r = 0; r < 8; r++) {
        const long row = m0 + wr * 32 + mt * 16 + kh * 8 + r;
        for (int nt = 0; nt < 4; nt++)
          outp[row * CD + ncol[nt]] = acc[mt][nt][r] + bv[nt];
      }
  }
}

// ------------------------------------------------------------------
// Attention: one block per (b,h,w). 5 waves (160 threads).
// Sb is pre-filled with earth-position bias (-1e30 in pad region) during
// staging; since padded Q/K rows are zero, the WMMA epilogue is a branch-free
// LDS accumulate. Then softmax (fp32) -> bf16 P -> P @ V(transposed in LDS).
// ------------------------------------------------------------------
__global__ __launch_bounds__(160) void attn_win_kernel(
    const __bf16* __restrict__ qb, const __bf16* __restrict__ kb,
    const __bf16* __restrict__ vb,
    const float* __restrict__ bias_table, const int* __restrict__ pos_index,
    __bf16* __restrict__ ao)
{
  __shared__ __align__(16) __bf16 Qs[80 * 32];
  __shared__ __align__(16) __bf16 Ks[80 * 32];
  __shared__ __align__(16) __bf16 Vt[32 * 96];   // transposed [d][m], m padded
  __shared__ __align__(16) float  Sb[80 * 80];
  __shared__ __align__(16) __bf16 Ps[80 * 96];   // K padded to 96 (3x32 steps)

  const int tid  = threadIdx.x;
  const int lane = tid & 31, wave = tid >> 5;    // wave 0..4
  const int kh = lane >> 4, ln = lane & 15;
  const int bid = blockIdx.x;
  const int w = bid % NW;
  const int h = (bid / NW) % NH;
  const int b = bid / (NW * NH);
  const long base = (((long)(b * NH + h) * NW + w) * NTK) * HD;

  const f32x8 zero = {0.f,0.f,0.f,0.f,0.f,0.f,0.f,0.f};

  // ---- stage Q,K (80x32 zero-padded), V^T (32x96), and bias into Sb ----
  for (int i = tid; i < 80 * 32; i += 160) {
    const int n = i >> 5, d = i & 31;
    __bf16 zq = (__bf16)0.f, zk = (__bf16)0.f;
    if (n < NTK) { zq = qb[base + n * HD + d]; zk = kb[base + n * HD + d]; }
    Qs[i] = zq; Ks[i] = zk;
  }
  for (int i = tid; i < 32 * 96; i += 160) {
    const int d = i / 96, m = i % 96;
    Vt[i] = (m < NTK) ? vb[base + m * HD + d] : (__bf16)0.f;
  }
#pragma unroll 4
  for (int i = tid; i < 80 * 80; i += 160) {     // independent gather chains
    const int n = i / 80, m = i % 80;
    float v = -1e30f;
    if (n < NTK && m < NTK)
      v = bias_table[(pos_index[n * NTK + m] * NW + w) * NH + h];
    Sb[i] = v;
  }
  __syncthreads();

  // ---- S += q k^T ; wave owns column tile j, B fragment hoisted ----
  {
    const int j = wave;
    const __bf16* bp = &Ks[(j * 16 + ln) * 32];
    bf16x16 bf = ld_frag2(bp + 8 * kh, bp + 16 + 8 * kh);
    const int m = j * 16 + ln;
    for (int i = 0; i < 5; i++) {
      const __bf16* ap = &Qs[(i * 16 + ln) * 32];
      bf16x16 af = ld_frag2(ap + 8 * kh, ap + 16 + 8 * kh);
      f32x8 s = wmma_bf16(af, bf, zero);
      float* sp = &Sb[(i * 16 + kh * 8) * 80 + m];
      for (int r = 0; r < 8; r++) sp[r * 80] += s[r];   // branch-free accumulate
    }
  }
  __syncthreads();

  // ---- row softmax (fp32), write bf16 P with zero padding ----
  if (tid < NTK) {
    float mx = -1e30f;
    for (int m = 0; m < NTK; m++) mx = fmaxf(mx, Sb[tid * 80 + m]);
    float sum = 0.f;
    for (int m = 0; m < NTK; m++) {
      const float e = __expf(Sb[tid * 80 + m] - mx);
      sum += e;
      Sb[tid * 80 + m] = e;
    }
    const float inv = 1.f / sum;
    for (int m = 0; m < NTK; m++) Ps[tid * 96 + m] = (__bf16)(Sb[tid * 80 + m] * inv);
    for (int m = NTK; m < 96; m++) Ps[tid * 96 + m] = (__bf16)0.f;
  } else {
    for (int i = tid - NTK; i < 8 * 96; i += (160 - NTK))
      Ps[(NTK + i / 96) * 96 + (i % 96)] = (__bf16)0.f;  // pad rows 72..79
  }
  __syncthreads();

  // ---- O = P @ V ; wave owns row tile; A fragments reused across col tiles ----
  {
    const int i = wave;
    bf16x16 af[3];
    for (int kk = 0; kk < 3; kk++) {
      const __bf16* ap = &Ps[(i * 16 + ln) * 96 + kk * 32];
      af[kk] = ld_frag2(ap + 8 * kh, ap + 16 + 8 * kh);
    }
    for (int ct = 0; ct < 2; ct++) {
      f32x8 acc = zero;
      for (int kk = 0; kk < 3; kk++) {
        const __bf16* bp = &Vt[(ct * 16 + ln) * 96 + kk * 32];
        bf16x16 bf = ld_frag2(bp + 8 * kh, bp + 16 + 8 * kh);
        acc = wmma_bf16(af[kk], bf, acc);
      }
      for (int r = 0; r < 8; r++) {
        const int n = i * 16 + kh * 8 + r;
        const int d = ct * 16 + ln;
        if (n < NTK) {
          const long t = (long)(b * NW + w) * NTK + n;
          ao[t * CD + h * HD + d] = (__bf16)acc[r];
        }
      }
    }
  }
}

// ------------------------------------------------------------------
extern "C" void kernel_launch(void* const* d_in, const int* in_sizes, int n_in,
                              void* d_out, int out_size, void* d_ws, size_t ws_size,
                              hipStream_t stream) {
  const float* x          = (const float*)d_in[0];
  const float* qkv_w      = (const float*)d_in[1];
  const float* qkv_b      = (const float*)d_in[2];
  const float* proj_w     = (const float*)d_in[3];
  const float* proj_b     = (const float*)d_in[4];
  const float* bias_table = (const float*)d_in[5];
  const int*   pos_index  = (const int*)d_in[6];
  float* out = (float*)d_out;

  char* ws = (char*)d_ws;
  size_t off = 0;
  auto alloc = [&](size_t bytes) -> void* {
    void* p = ws + off;
    off = (off + bytes + 255) & ~(size_t)255;
    return p;
  };

  const size_t TC = (size_t)TTOT * CD;              // 26,542,080 elems
  __bf16* xb    = (__bf16*)alloc(TC * 2);           // reused as attention out
  __bf16* wqkv  = (__bf16*)alloc((size_t)O3 * CD * 2);
  __bf16* wproj = (__bf16*)alloc((size_t)CD * CD * 2);
  __bf16* qbuf  = (__bf16*)alloc(TC * 2);           // (b,h,w,n,d)
  __bf16* kbuf  = (__bf16*)alloc(TC * 2);
  __bf16* vbuf  = (__bf16*)alloc(TC * 2);
  __bf16* ao    = xb;  // x dead after the QKV GEMM (stream-ordered)

  // 1) precision demotion
  {
    int n = (int)TC;
    cvt_f32_to_bf16<<<(n + 255) / 256, 256, 0, stream>>>(x, xb, n);
    n = O3 * CD;
    cvt_f32_to_bf16<<<(n + 255) / 256, 256, 0, stream>>>(qkv_w, wqkv, n);
    n = CD * CD;
    cvt_f32_to_bf16<<<(n + 255) / 256, 256, 0, stream>>>(proj_w, wproj, n);
  }

  // 2) QKV projection GEMM: 69120 x 1152 x 384
  gemm_bf16_128<0><<<dim3(TTOT / 128, O3 / 128), 256, 0, stream>>>(
      xb, wqkv, qkv_b, qbuf, kbuf, vbuf, nullptr);

  // 3) windowed attention: one block per (b,h,w)
  attn_win_kernel<<<B_SZ * NH * NW, 160, 0, stream>>>(
      qbuf, kbuf, vbuf, bias_table, pos_index, ao);

  // 4) output projection GEMM: 69120 x 384 x 384 (+proj_b), fp32 out
  gemm_bf16_128<1><<<dim3(TTOT / 128, CD / 128), 256, 0, stream>>>(
      ao, wproj, proj_b, nullptr, nullptr, nullptr, out);
}
